// CustomLSTM_89962384982894
// MI455X (gfx1250) — matrix operand order repaired and checked by reading, not compiled, not run on hardware
//
#include <hip/hip_runtime.h>
#include <hip/hip_bf16.h>
#include <math.h>
#include <stdint.h>

#ifndef __has_builtin
#define __has_builtin(x) 0
#endif

// ---------------------------------------------------------------------------
// CDNA5 (gfx1250, wave32) types
// ---------------------------------------------------------------------------
typedef __bf16 v16bf __attribute__((ext_vector_type(16)));
typedef float  v8f   __attribute__((ext_vector_type(8)));
typedef unsigned int u32x4 __attribute__((ext_vector_type(4)));
typedef int          i32x4 __attribute__((ext_vector_type(4)));
typedef int          i32x8 __attribute__((ext_vector_type(8)));

union FragB16 { v16bf v; unsigned int u[8]; };

__device__ __forceinline__ unsigned short f2bf(float f) {
    unsigned int x = __float_as_uint(f);
    x += 0x7FFFu + ((x >> 16) & 1u);          // round-to-nearest-even
    return (unsigned short)(x >> 16);
}
__device__ __forceinline__ float bf2f(unsigned short h) {
    return __uint_as_float(((unsigned int)h) << 16);
}
__device__ __forceinline__ v8f zero8() {
    v8f z = {0.f, 0.f, 0.f, 0.f, 0.f, 0.f, 0.f, 0.f};
    return z;
}

#define NWG       8          // persistent workgroups in the scan kernel
#define ICP       66         // padded ic stride in conv1-out LDS tile (bank-conflict pad)

// LDS layout for conv kernel (bytes):
//   r0: max(input 3*68*68*4 = 55488, B2 frags 144*256*4 = 147456)  -> 147456
//   r1: conv1-out tile 33*33*66 u16 = 143748 -> pad 143760
//   r2: B1 frags 4*256*4 = 4096
//   r3: pool[128] f32 = 512
#define SMEM_CONV (147456 + 143760 + 4096 + 512)

// LDS layout for scan kernel:
//   lfc 50*1536*2=153600 | lfca 64*400*2=51200 | act 1600 | xbuf 6144 | llog 1600 | red 2048
#define SMEM_SCAN (153600 + 51200 + 1600 + 6144 + 1600 + 2048)

// ---------------------------------------------------------------------------
// Tensor Data Mover: one-shot 2D tile (rows x width dwords) global -> LDS.
// Issued by a single wave; tracked by TENSORcnt.  Descriptor per ISA 8.3/8.4.
// ---------------------------------------------------------------------------
#if __has_builtin(__builtin_amdgcn_tensor_load_to_lds)
#define HAS_TDM 1
__device__ __forceinline__ void tdm_load_2d(unsigned lds_off, const void* gptr,
                                            unsigned width_dw, unsigned rows) {
    unsigned long long ga = (unsigned long long)(uintptr_t)gptr;
    u32x4 g0;
    g0[0] = 1u;                                             // count=1 (valid user D#)
    g0[1] = lds_off;                                        // lds_addr (bytes)
    g0[2] = (unsigned)(ga & 0xFFFFFFFFull);                 // global_addr[31:0]
    g0[3] = (unsigned)((ga >> 32) & 0x1FFFFFFull)           // global_addr[56:32]
          | (2u << 30);                                     // type=2 ("image")
    i32x8 g1;
    g1[0] = 0x20000;                                        // data_size=2 -> 4 bytes
    g1[1] = (int)(width_dw << 16);                          // tensor_dim0[15:0]
    g1[2] = (int)(rows << 16);                              // tensor_dim1[15:0]
    g1[3] = (int)(width_dw << 16);                          // tile_dim0
    g1[4] = (int)rows;                                      // tile_dim1
    g1[5] = (int)width_dw;                                  // tensor_dim0_stride[31:0]
    g1[6] = 0;
    g1[7] = 0;
    i32x4 gz4 = {0, 0, 0, 0};
#if __has_include(<hip/amd_detail/amd_gfx1250_TDM.h>)
    i32x8 gz8 = {0, 0, 0, 0, 0, 0, 0, 0};                   // clang-23 / therock: 6-arg form
    __builtin_amdgcn_tensor_load_to_lds(g0, g1, gz4, gz4, gz8, 0);
#else
    __builtin_amdgcn_tensor_load_to_lds(g0, g1, gz4, gz4, 0);  // ROCm 7.2: 5-arg form
#endif
}
#endif

// ---------------------------------------------------------------------------
// Kernel 0: pack conv weights into bf16 WMMA B-fragments (exact wave32 register
// layout: [frag][lane][8 dwords], K-order = (ky*3+kx)*IC + ic), zero feats+sync.
// ---------------------------------------------------------------------------
__global__ void pack_init_kernel(const float* __restrict__ w1,
                                 const float* __restrict__ w2,
                                 unsigned int* __restrict__ b1f,
                                 unsigned int* __restrict__ b2f,
                                 float* __restrict__ feats,
                                 unsigned int* __restrict__ syncbuf) {
    int tid = blockIdx.x * blockDim.x + threadIdx.x;
    if (tid < 1024) {                                   // conv1 B frags: 4 n-tiles x 1 k-step
        int f = tid >> 8;                               // n-tile
        int r = tid & 255;
        int lane = r >> 3, v = r & 7;
        int klocal = ((v & 4) << 2) + (lane >> 4) * 8 + (v & 3) * 2;  // B layout: lanes16-31 hold K=8..15,24..31
        int oc = f * 16 + (lane & 15);
        unsigned int pack = 0;
        for (int h = 0; h < 2; ++h) {
            int k = klocal + h;                         // k = (ky*3+kx)*3 + ic, pad 27..31 = 0
            float val = 0.f;
            if (k < 27) {
                int p = k / 3, ic = k % 3, ky = p / 3, kx = p % 3;
                val = w1[((oc * 3 + ic) * 3 + ky) * 3 + kx];
            }
            pack |= ((unsigned int)f2bf(val)) << (16 * h);
        }
        b1f[tid] = pack;
    } else if (tid < 1024 + 144 * 256) {                // conv2 B frags: 8 n-tiles x 18 k-steps
        int e = tid - 1024;
        int f = e >> 8;
        int r = e & 255;
        int lane = r >> 3, v = r & 7;
        int nt = f / 18, kk = f % 18;
        int klocal = ((v & 4) << 2) + (lane >> 4) * 8 + (v & 3) * 2;
        int oc = nt * 16 + (lane & 15);
        unsigned int pack = 0;
        for (int h = 0; h < 2; ++h) {
            int k = kk * 32 + klocal + h;               // k = (ky*3+kx)*64 + ic
            int p = k >> 6, ic = k & 63, ky = p / 3, kx = p % 3;
            float val = w2[((oc * 64 + ic) * 3 + ky) * 3 + kx];
            pack |= ((unsigned int)f2bf(val)) << (16 * h);
        }
        b2f[e] = pack;
    } else {
        int e = tid - 1024 - 144 * 256;
        if (e < 128 * 128)            feats[e] = 0.f;   // per-frame pooled feature accumulator
        else if (e < 128 * 128 + 2)   syncbuf[e - 128 * 128] = 0u;  // grid barrier state
    }
}

// ---------------------------------------------------------------------------
// Kernel 1: fused conv1 -> ReLU -> conv2 -> ReLU -> avg-pool, all via WMMA.
// Grid: 2048 blocks (128 frames x 4x4 tiles of the 64x64 conv2 output),
// 256 threads = 8 wave32s.  Everything stays in LDS between the convs.
// ---------------------------------------------------------------------------
__global__ __launch_bounds__(256, 1)
void conv_fused_kernel(const float* __restrict__ video,
                       const float* __restrict__ b1,
                       const float* __restrict__ b2,
                       const unsigned int* __restrict__ b1f,
                       const unsigned int* __restrict__ b2f,
                       float* __restrict__ feats) {
    extern __shared__ char smem[];
    float*          inp  = (float*)smem;                                   // [3][68][68]
    unsigned int*   lb2  = (unsigned int*)smem;                            // overlays inp later
    unsigned short* c1   = (unsigned short*)(smem + 147456);               // [33*33][ICP]
    unsigned int*   lb1  = (unsigned int*)(smem + 147456 + 143760);        // [4*256]
    float*          pool = (float*)(smem + 147456 + 143760 + 4096);        // [128]

    const int tid   = threadIdx.x;
    const int lane  = tid & 31;
    const int wave  = tid >> 5;
    const int frame = blockIdx.x >> 4;
    const int tile  = blockIdx.x & 15;
    const int ty    = tile >> 2, tx = tile & 3;

    // ---- phase A: stage input tile (with conv1 zero-pad), zero c1, load B1 frags
    for (int e = tid; e < 3 * 68 * 68; e += 256) {
        int ic = e / (68 * 68);
        int rem = e % (68 * 68);
        int li = rem / 68, lj = rem % 68;
        float v = 0.f;
        if (li < 67 && lj < 67) {
            int gy = 64 * ty - 3 + li;
            int gx = 64 * tx - 3 + lj;
            if (gy >= 0 && gy < 256 && gx >= 0 && gx < 256)
                v = video[((frame * 3 + ic) * 256 + gy) * 256 + gx];
        }
        inp[e] = v;
    }
    for (int e = tid; e < (1089 * ICP) / 2; e += 256) ((unsigned int*)c1)[e] = 0u;
    for (int e = tid; e < 1024; e += 256) lb1[e] = b1f[e];
    if (tid < 128) pool[tid] = 0.f;
    if (tid == 0) __builtin_prefetch(b2f, 0, 1);        // global_prefetch_b8 (L2 warm-up)
    __syncthreads();

    // ---- phase B: conv1 via WMMA (M=1089 px padded to 69 tiles, N=64, K=27->32)
    for (int mt = wave; mt < 69; mt += 8) {
        const int mrow = lane & 15, kh = lane >> 4;
        const int pix  = mt * 16 + mrow;
        const bool pvalid = pix < 1089;
        const int l1y = pvalid ? pix / 33 : 0;
        const int l1x = pvalid ? pix % 33 : 0;
        FragB16 A;
#pragma unroll
        for (int v8 = 0; v8 < 8; ++v8) {
            int klocal = ((v8 & 4) << 2) + kh * 8 + (v8 & 3) * 2;  // ISA 16-bit A layout
            unsigned int pk = 0;
#pragma unroll
            for (int h = 0; h < 2; ++h) {
                int k = klocal + h;
                float val = 0.f;
                if (pvalid && k < 27) {
                    int p = k / 3, ic = k % 3, ky = p / 3, kx = p % 3;
                    val = inp[(ic * 68 + (2 * l1y + ky)) * 68 + (2 * l1x + kx)];
                }
                pk |= ((unsigned int)f2bf(val)) << (16 * h);
            }
            A.u[v8] = pk;
        }
        // hoisted store coordinates (shared by all 4 n-tiles)
        int  saddr[8];
        bool sok[8];
#pragma unroll
        for (int j = 0; j < 8; ++j) {
            int M  = j + ((lane >> 4) << 3);
            int pp = mt * 16 + M;
            bool ok = pp < 1089;
            int ry = ok ? pp / 33 : 0;
            int rx = ok ? pp % 33 : 0;
            // conv2 zero-padding ring: drop conv1-out row/col -1
            sok[j]   = ok && (32 * ty - 1 + ry >= 0) && (32 * tx - 1 + rx >= 0);
            saddr[j] = (ry * 33 + rx) * ICP;
        }
#pragma unroll
        for (int nt = 0; nt < 4; ++nt) {
            FragB16 B;
            const uint4* bp = (const uint4*)(lb1 + nt * 256 + lane * 8);
            *(uint4*)&B.u[0] = bp[0];
            *(uint4*)&B.u[4] = bp[1];
            v8f acc = zero8();
            acc = __builtin_amdgcn_wmma_f32_16x16x32_bf16(
                false, A.v, false, B.v, (short)0, acc, false, false);
            const int oc   = nt * 16 + (lane & 15);
            const float bs = b1[oc];
#pragma unroll
            for (int j = 0; j < 8; ++j) {
                if (sok[j]) {
                    float o = acc[j] + bs;
                    o = o > 0.f ? o : 0.f;               // ReLU
                    c1[saddr[j] + oc] = f2bf(o);
                }
            }
        }
    }
    __syncthreads();

    // ---- phase C: stage conv2 B-fragments into LDS (overlays input tile)
#ifdef HAS_TDM
    if (wave == 0)
        tdm_load_2d((unsigned)(uintptr_t)lb2, b2f, /*width_dw=*/4096, /*rows=*/9);
    __builtin_amdgcn_s_wait_tensorcnt((short)0);
#else
    for (int e = tid; e < 144 * 256; e += 256) lb2[e] = b2f[e];
#endif
    __syncthreads();

    // ---- phase D: conv2 via WMMA. Wave handles 2 M-tiles x 8 N-tiles, K=576 (18 steps)
    {
        const int mrow = lane & 15, kh = lane >> 4;
        const int mtA = wave * 2, mtB = wave * 2 + 1;
        const int pixA = mtA * 16 + mrow, pyA = pixA >> 4, pxA = pixA & 15;
        const int pixB = mtB * 16 + mrow, pyB = pixB >> 4, pxB = pixB & 15;
        v8f acc0[8], acc1[8];
#pragma unroll
        for (int n = 0; n < 8; ++n) { acc0[n] = zero8(); acc1[n] = zero8(); }

        for (int kk = 0; kk < 18; ++kk) {
            const int kykx = kk >> 1, icb = (kk & 1) * 32;
            const int ky = kykx / 3, kx = kykx % 3;
            const unsigned short* baseA0 =
                c1 + ((2 * pyA + ky) * 33 + (2 * pxA + kx)) * ICP + icb;
            const unsigned short* baseA1 =
                c1 + ((2 * pyB + ky) * 33 + (2 * pxB + kx)) * ICP + icb;
            FragB16 A0, A1;
#pragma unroll
            for (int v8 = 0; v8 < 8; ++v8) {
                int klocal = ((v8 & 4) << 2) + kh * 8 + (v8 & 3) * 2;
                A0.u[v8] = *(const unsigned int*)(baseA0 + klocal);  // 2 contiguous ic's
                A1.u[v8] = *(const unsigned int*)(baseA1 + klocal);
            }
#pragma unroll
            for (int nt = 0; nt < 8; ++nt) {
                FragB16 B;
                const uint4* bp = (const uint4*)(lb2 + (nt * 18 + kk) * 256 + lane * 8);
                *(uint4*)&B.u[0] = bp[0];
                *(uint4*)&B.u[4] = bp[1];
                acc0[nt] = __builtin_amdgcn_wmma_f32_16x16x32_bf16(
                    false, A0.v, false, B.v, (short)0, acc0[nt], false, false);
                acc1[nt] = __builtin_amdgcn_wmma_f32_16x16x32_bf16(
                    false, A1.v, false, B.v, (short)0, acc1[nt], false, false);
            }
        }
        // epilogue: bias + ReLU + partial avg-pool into LDS pool[128]
#pragma unroll
        for (int nt = 0; nt < 8; ++nt) {
            int oc = nt * 16 + (lane & 15);
            float bs = b2[oc];
            float s = 0.f;
#pragma unroll
            for (int j = 0; j < 8; ++j) {
                float a0 = acc0[nt][j] + bs; s += a0 > 0.f ? a0 : 0.f;
                float a1 = acc1[nt][j] + bs; s += a1 > 0.f ? a1 : 0.f;
            }
            atomicAdd(&pool[oc], s);
        }
    }
    __syncthreads();
    if (tid < 128) atomicAdd(&feats[frame * 128 + tid], pool[tid]);
}

// ---------------------------------------------------------------------------
// Kernel 2: encoded-video branch: ev = relu([14,2048] @ fce_w.T + b) -> [896]
// ---------------------------------------------------------------------------
__global__ void fce_kernel(const float* __restrict__ enc,
                           const float* __restrict__ w,
                           const float* __restrict__ b,
                           float* __restrict__ ev) {
    int i = threadIdx.x;
    if (i < 896) {
        int n = i >> 6, o = i & 63;
        const float4* e4 = (const float4*)(enc + n * 2048);
        const float4* w4 = (const float4*)(w + o * 2048);
        float s = 0.f;
        for (int d = 0; d < 512; ++d) {
            float4 a = e4[d], c = w4[d];
            s += a.x * c.x + a.y * c.y + a.z * c.z + a.w * c.w;
        }
        s += b[o];
        ev[i] = s > 0.f ? s : 0.f;
    }
}

// ---------------------------------------------------------------------------
// Grid-wide sense-reversing barrier (agent scope)
// ---------------------------------------------------------------------------
__device__ __forceinline__ void gsync(unsigned int* cnt, unsigned int* gen) {
    __threadfence();
    __syncthreads();
    if (threadIdx.x == 0) {
        unsigned int g = __hip_atomic_load(gen, __ATOMIC_RELAXED, __HIP_MEMORY_SCOPE_AGENT);
        unsigned int prev =
            __hip_atomic_fetch_add(cnt, 1u, __ATOMIC_ACQ_REL, __HIP_MEMORY_SCOPE_AGENT);
        if (prev == NWG - 1) {
            __hip_atomic_store(cnt, 0u, __ATOMIC_RELAXED, __HIP_MEMORY_SCOPE_AGENT);
            __hip_atomic_fetch_add(gen, 1u, __ATOMIC_RELEASE, __HIP_MEMORY_SCOPE_AGENT);
        } else {
            while (__hip_atomic_load(gen, __ATOMIC_ACQUIRE, __HIP_MEMORY_SCOPE_AGENT) == g)
                __builtin_amdgcn_s_sleep(8);
        }
    }
    __syncthreads();
}

// ---------------------------------------------------------------------------
// Kernel 3: the serial 128-step scan. The LSTM cell is dead code w.r.t. the
// output, so each step is: a = relu(fca(act)); x=[feat,ev,a]; out=softmax(fc(x)).
// 8 persistent WGs keep their bf16 weight slices resident in LDS; 2 grid syncs
// per step; softmax recomputed per-WG (removes a 3rd sync).
// ---------------------------------------------------------------------------
__global__ __launch_bounds__(512, 1)
void scan_kernel(const float* __restrict__ feats,
                 const float* __restrict__ ev,
                 const float* __restrict__ action,
                 const float* __restrict__ fca_w, const float* __restrict__ fca_b,
                 const float* __restrict__ fc_w,  const float* __restrict__ fc_b,
                 float* __restrict__ ga, float* __restrict__ glog,
                 unsigned int* __restrict__ syncbuf,
                 float* __restrict__ out) {
    extern __shared__ char smem[];
    unsigned short* lfc  = (unsigned short*)smem;                    // [50][1536] bf16
    unsigned short* lfca = (unsigned short*)(smem + 153600);         // [64][400]  bf16
    float* act  = (float*)(smem + 153600 + 51200);                   // [400]
    float* xbuf = (float*)(smem + 153600 + 51200 + 1600);            // [1536]
    float* llog = (float*)(smem + 153600 + 51200 + 1600 + 6144);     // [400]
    float* red  = (float*)(smem + 153600 + 51200 + 1600 + 6144 + 1600); // [512]

    const int tid = threadIdx.x;
    const int wg  = blockIdx.x;
    const int fcr0 = wg * 50;   // fc rows owned by this WG
    const int far0 = wg * 64;   // fca rows owned by this WG

    for (int e = tid; e < 50 * 1536; e += 512) lfc[e]  = f2bf(fc_w[fcr0 * 1536 + e]);
    for (int e = tid; e < 64 * 400;  e += 512) lfca[e] = f2bf(fca_w[far0 * 400 + e]);
    if (tid < 400) act[tid] = action[tid];
    __syncthreads();

    unsigned int* cnt = syncbuf;
    unsigned int* gen = syncbuf + 1;

    for (int t = 0; t < 128; ++t) {
        // phase 1: a-slice (64 rows x 8 lanes each)
        {
            int r = tid >> 3, sub = tid & 7;
            float p = 0.f;
            for (int j = sub; j < 400; j += 8) p += bf2f(lfca[r * 400 + j]) * act[j];
            p += __shfl_xor(p, 1, 8);
            p += __shfl_xor(p, 2, 8);
            p += __shfl_xor(p, 4, 8);
            if (sub == 0) {
                float v = p + fca_b[far0 + r];
                ga[far0 + r] = v > 0.f ? v : 0.f;
            }
        }
        gsync(cnt, gen);

        // assemble x = [feat_t/4096 (128), ev (896), a (512)]
        for (int j = tid; j < 1536; j += 512) {
            float v;
            if (j < 128)       v = feats[t * 128 + j] * (1.0f / 4096.0f);
            else if (j < 1024) v = ev[j - 128];
            else               v = ga[j - 1024];
            xbuf[j] = v;
        }
        __syncthreads();

        // phase 2: logit slice (50 rows x 8 lanes each)
        if (tid < 400) {
            int r = tid >> 3, sub = tid & 7;
            float p = 0.f;
            for (int j = sub; j < 1536; j += 8) p += bf2f(lfc[r * 1536 + j]) * xbuf[j];
            p += __shfl_xor(p, 1, 8);
            p += __shfl_xor(p, 2, 8);
            p += __shfl_xor(p, 4, 8);
            if (sub == 0) glog[fcr0 + r] = p + fc_b[fcr0 + r];
        }
        gsync(cnt, gen);

        // softmax (every WG redundantly -> act ready locally, no 3rd sync)
        float li = -3.4e38f;
        if (tid < 400) { li = glog[tid]; llog[tid] = li; }
        red[tid] = li;
        __syncthreads();
        for (int s = 256; s > 0; s >>= 1) {
            if (tid < s) { float o = red[tid + s]; if (o > red[tid]) red[tid] = o; }
            __syncthreads();
        }
        float mx = red[0];
        __syncthreads();
        float e = 0.f;
        if (tid < 400) e = __expf(llog[tid] - mx);
        red[tid] = e;
        __syncthreads();
        for (int s = 256; s > 0; s >>= 1) {
            if (tid < s) red[tid] += red[tid + s];
            __syncthreads();
        }
        float inv = 1.0f / red[0];
        __syncthreads();
        if (tid < 400) {
            float o = e * inv;
            act[tid] = o;
            if (wg == 0) out[t * 400 + tid] = o;
        }
        __syncthreads();
    }
}

// ---------------------------------------------------------------------------
extern "C" void kernel_launch(void* const* d_in, const int* in_sizes, int n_in,
                              void* d_out, int out_size, void* d_ws, size_t ws_size,
                              hipStream_t stream) {
    (void)in_sizes; (void)n_in; (void)out_size; (void)ws_size;
    const float* enc    = (const float*)d_in[0];
    const float* video  = (const float*)d_in[1];
    const float* action = (const float*)d_in[2];
    const float* w1     = (const float*)d_in[3];
    const float* b1     = (const float*)d_in[4];
    const float* w2     = (const float*)d_in[5];
    const float* b2     = (const float*)d_in[6];
    const float* fce_w  = (const float*)d_in[7];
    const float* fce_b  = (const float*)d_in[8];
    const float* fca_w  = (const float*)d_in[9];
    const float* fca_b  = (const float*)d_in[10];
    const float* fc_w   = (const float*)d_in[11];
    const float* fc_b   = (const float*)d_in[12];
    float* out = (float*)d_out;

    char* ws = (char*)d_ws;
    unsigned int* b1f   = (unsigned int*)(ws + 0);       //   4 KB conv1 B-frags
    unsigned int* b2f   = (unsigned int*)(ws + 4096);    // 144 KB conv2 B-frags
    float*        feats = (float*)(ws + 151552);         //  64 KB pooled features
    float*        ev    = (float*)(ws + 217088);         // 3.5 KB encoded-video
    float*        ga    = (float*)(ws + 220672);         //   2 KB a-vector
    float*        glog  = (float*)(ws + 222720);         // 1.6 KB logits
    unsigned int* syncb = (unsigned int*)(ws + 224320);  //   8 B barrier state

    (void)hipFuncSetAttribute((const void*)conv_fused_kernel,
                              hipFuncAttributeMaxDynamicSharedMemorySize, SMEM_CONV);
    (void)hipFuncSetAttribute((const void*)scan_kernel,
                              hipFuncAttributeMaxDynamicSharedMemorySize, SMEM_SCAN);

    pack_init_kernel<<<213, 256, 0, stream>>>(w1, w2, b1f, b2f, feats, syncb);
    conv_fused_kernel<<<2048, 256, SMEM_CONV, stream>>>(video, b1, b2, b1f, b2f, feats);
    fce_kernel<<<1, 896, 0, stream>>>(enc, fce_w, fce_b, ev);
    scan_kernel<<<NWG, 512, SMEM_SCAN, stream>>>(feats, ev, action,
                                                 fca_w, fca_b, fc_w, fc_b,
                                                 ga, glog, syncb, out);
}
